// Ourmamba_71966472012603
// MI455X (gfx1250) — compile-verified, matrix operand order checked by reading, not run on hardware
//
#include <hip/hip_runtime.h>

typedef __attribute__((ext_vector_type(16))) __bf16 v16bf;
typedef __attribute__((ext_vector_type(8)))  __bf16 v8bf;
typedef __attribute__((ext_vector_type(8)))  float  v8f;
typedef __attribute__((ext_vector_type(4)))  unsigned int v4u;
typedef __attribute__((ext_vector_type(8)))  int v8i;
typedef __attribute__((ext_vector_type(4)))  int v4i;

#define DM     1024
#define DIN    2048
#define LSEQ   2048
#define TTOK   4096   // B * L
#define DSTATE 16
#define RNK    64
#define XDBL   96     // dt_rank + 2*d_state

#if defined(__gfx1250__) && __has_builtin(__builtin_amdgcn_tensor_load_to_lds) && __has_builtin(__builtin_amdgcn_s_wait_tensorcnt)
#define HAVE_TDM 1
#endif

__device__ __forceinline__ v16bf cat16(v8bf lo, v8bf hi) {
  return __builtin_shufflevector(lo, hi, 0,1,2,3,4,5,6,7,8,9,10,11,12,13,14,15);
}
__device__ __forceinline__ float siluf(float v)     { return v / (1.0f + __expf(-v)); }
__device__ __forceinline__ float softplusf(float v) { return (v > 20.0f) ? v : __logf(1.0f + __expf(v)); }

// ---------------- f32 -> bf16 convert ----------------
__global__ void k_cvt_bf16(const float* __restrict__ in, __bf16* __restrict__ out, int n) {
  int i = blockIdx.x * 256 + threadIdx.x;
  if (i < n) out[i] = (__bf16)in[i];
}

#ifdef HAVE_TDM
// Issue one TDM 2D tile load: 32 (k) x 64 (rows) bf16 elements from a row-major
// [rows x strideK] tensor into LDS at lds_off, padded +16B per 64B row so the
// LDS image has an 80-byte row stride (matches As/Ws [64][40] layout).
__device__ __forceinline__ void tdm_load_tile(const __bf16* gptr, unsigned lds_off,
                                              int tensK, int tensRows, int strideK)
{
  unsigned long long ga = (unsigned long long)(size_t)gptr;
  v4u g0;
  g0[0] = 1u;                                     // count=1, user descriptor
  g0[1] = lds_off;                                // lds_addr (bytes)
  g0[2] = (unsigned)(ga & 0xFFFFFFFFu);           // global_addr[31:0]
  g0[3] = (unsigned)((ga >> 32) & 0x01FFFFFFu)    // global_addr[56:32]
        | 0x80000000u;                            // type=2 ("image")
  v8i g1;
  g1[0] = (int)((1u << 16)      // data_size = 2 bytes
              | (1u << 20)      // pad_enable
              | (3u << 22)      // pad_interval: 16 DWORDs (one 64B tile row)
              | (3u << 25));    // pad_amount: 4 DWORDs (16B) -> 80B row stride
  g1[1] = (int)(((unsigned)tensK & 0xFFFFu) << 16);                       // tensor_dim0[15:0]
  g1[2] = (int)((((unsigned)tensK >> 16) & 0xFFFFu)
              | (((unsigned)tensRows & 0xFFFFu) << 16));                  // dim0[31:16] | dim1[15:0]
  g1[3] = (int)((((unsigned)tensRows >> 16) & 0xFFFFu) | (32u << 16));    // dim1[31:16] | tile_dim0=32
  g1[4] = (int)(64u);                                                     // tile_dim1=64, tile_dim2=0
  g1[5] = (int)(unsigned)strideK;                                         // tensor_dim0_stride[31:0]
  g1[6] = 0;                                                              // stride0[47:32] | stride1[15:0]
  g1[7] = 0;
  v4i z4; z4[0] = 0; z4[1] = 0; z4[2] = 0; z4[3] = 0;
#if __clang_major__ >= 23
  v8i z8; z8[0]=0; z8[1]=0; z8[2]=0; z8[3]=0; z8[4]=0; z8[5]=0; z8[6]=0; z8[7]=0;
  __builtin_amdgcn_tensor_load_to_lds(g0, g1, z4, z4, z8, 0);
#else
  __builtin_amdgcn_tensor_load_to_lds(g0, g1, z4, z4, 0);
#endif
}
#endif

// ---------------- generic WMMA GEMM: C[m,n] = sum_k A[m,k]*W[n,k] ----------------
// A: [M,K] bf16 row-major (M = gridDim.y*64). W: [N,K] bf16 row-major.
// mode 0: C0[row*N+col] = v
// mode 1: col<DIN ? C0[row*DIN+col]=v : C1[row*DIN+col-DIN]=v   (in_proj split x/z)
// mode 2: C0[row*N+col] = softplus(v + bias[col])               (dt_proj)
__global__ __launch_bounds__(256)
void k_wmma_gemm(const __bf16* __restrict__ A, const __bf16* __restrict__ W,
                 const float* __restrict__ bias, float* __restrict__ C0,
                 float* __restrict__ C1, int N, int K, int mode)
{
  const int tid  = threadIdx.x;
  const int lane = tid & 31;
  const int w    = tid >> 5;
  const int wm   = w & 3;          // 4 wave rows (M)
  const int wn   = w >> 2;         // 2 wave cols (N), 32 cols each
  const int hi   = lane >> 4;
  const int ln   = lane & 15;
  const int m0   = blockIdx.y * 64;
  const int n0   = blockIdx.x * 64;
  const int k0   = hi * 8;

  v8f acc0, acc1;
#pragma unroll
  for (int i = 0; i < 8; ++i) { acc0[i] = 0.0f; acc1[i] = 0.0f; }

#ifdef HAVE_TDM
  // --- double-buffered tiles filled by the Tensor Data Mover ---
  __shared__ __bf16 As[2][64][40];
  __shared__ __bf16 Ws[2][64][40];
  const unsigned ldsA0 = (unsigned)(size_t)&As[0][0][0];  // LDS aperture: addr[31:0] = LDS offset
  const unsigned ldsA1 = (unsigned)(size_t)&As[1][0][0];
  const unsigned ldsW0 = (unsigned)(size_t)&Ws[0][0][0];
  const unsigned ldsW1 = (unsigned)(size_t)&Ws[1][0][0];
  const int rowsW = (N - n0) > 64 ? 64 : (N - n0);

  if (tid < 32) {  // wave 0 drives the DMA (EXEC ignored by TDM, per-wave TENSORcnt)
    tdm_load_tile(A + (size_t)m0 * K, ldsA0, K, 64, K);
    tdm_load_tile(W + (size_t)n0 * K, ldsW0, K, rowsW, K);
  }
  for (int kb = 0; kb < K; kb += 32) {
    const int p = (kb >> 5) & 1;
    if (tid < 32) __builtin_amdgcn_s_wait_tensorcnt((short)0);
    __syncthreads();                       // buffer p ready for everyone
    if (tid < 32 && (kb + 32) < K) {       // start DMA of next tile into buffer 1-p
      tdm_load_tile(A + (size_t)m0 * K + kb + 32, p ? ldsA0 : ldsA1, K - kb - 32, 64, K);
      tdm_load_tile(W + (size_t)n0 * K + kb + 32, p ? ldsW0 : ldsW1, K - kb - 32, rowsW, K);
    }
    const int mr = wm * 16 + ln;
    v16bf am = cat16(*(const v8bf*)&As[p][mr][k0], *(const v8bf*)&As[p][mr][k0 + 16]);
    const int nc = wn * 32 + ln;
    v16bf b0 = cat16(*(const v8bf*)&Ws[p][nc][k0],      *(const v8bf*)&Ws[p][nc][k0 + 16]);
    v16bf b1 = cat16(*(const v8bf*)&Ws[p][nc + 16][k0], *(const v8bf*)&Ws[p][nc + 16][k0 + 16]);
    acc0 = __builtin_amdgcn_wmma_f32_16x16x32_bf16(false, am, false, b0, (short)0, acc0, false, false);
    acc1 = __builtin_amdgcn_wmma_f32_16x16x32_bf16(false, am, false, b1, (short)0, acc1, false, false);
    __syncthreads();                       // all reads of buffer p done before it is overwritten
  }
#else
  // --- fallback: synchronous VGPR-staged tiles ---
  __shared__ __bf16 As[64][40];
  __shared__ __bf16 Ws[64][40];
  const int ldRow = tid >> 2;          // 0..63
  const int ldK   = (tid & 3) * 8;     // 0,8,16,24
  const int aRow = m0 + ldRow;
  const int wRow = n0 + ldRow;
  for (int kb = 0; kb < K; kb += 32) {
    v8bf av = *(const v8bf*)(A + (size_t)aRow * K + kb + ldK);
    v8bf wv;
#pragma unroll
    for (int i = 0; i < 8; ++i) wv[i] = (__bf16)0.0f;
    if (wRow < N) wv = *(const v8bf*)(W + (size_t)wRow * K + kb + ldK);
    *(v8bf*)&As[ldRow][ldK] = av;
    *(v8bf*)&Ws[ldRow][ldK] = wv;
    __syncthreads();
    if (kb + 32 < K) {
      __builtin_prefetch(A + (size_t)aRow * K + kb + 32 + ldK, 0, 1);
      if (wRow < N) __builtin_prefetch(W + (size_t)wRow * K + kb + 32 + ldK, 0, 1);
    }
    const int mr = wm * 16 + ln;
    v16bf am = cat16(*(const v8bf*)&As[mr][k0], *(const v8bf*)&As[mr][k0 + 16]);
    const int nc = wn * 32 + ln;
    v16bf b0 = cat16(*(const v8bf*)&Ws[nc][k0],      *(const v8bf*)&Ws[nc][k0 + 16]);
    v16bf b1 = cat16(*(const v8bf*)&Ws[nc + 16][k0], *(const v8bf*)&Ws[nc + 16][k0 + 16]);
    acc0 = __builtin_amdgcn_wmma_f32_16x16x32_bf16(false, am, false, b0, (short)0, acc0, false, false);
    acc1 = __builtin_amdgcn_wmma_f32_16x16x32_bf16(false, am, false, b1, (short)0, acc1, false, false);
    __syncthreads();
  }
#endif

  const int colBase = n0 + wn * 32 + ln;
#pragma unroll
  for (int r = 0; r < 8; ++r) {
    const int row = m0 + wm * 16 + r + hi * 8;
#pragma unroll
    for (int t = 0; t < 2; ++t) {
      const int col = colBase + t * 16;
      const float v = (t == 0) ? acc0[r] : acc1[r];
      if (mode == 0) {
        if (col < N) C0[(size_t)row * N + col] = v;
      } else if (mode == 1) {
        if (col < DIN) C0[(size_t)row * DIN + col] = v;
        else           C1[(size_t)row * DIN + (col - DIN)] = v;
      } else {
        if (col < N) C0[(size_t)row * N + col] = softplusf(v + bias[col]);
      }
    }
  }
}

// ---------------- causal depthwise conv (d_conv=4) + bias + SiLU -> bf16 ----------------
__global__ void k_conv_silu(const float* __restrict__ x_raw, const float* __restrict__ conv_w,
                            const float* __restrict__ conv_b, __bf16* __restrict__ x_act)
{
  int idx = blockIdx.x * 256 + threadIdx.x;
  if (idx >= TTOK * DIN) return;
  const int d = idx & (DIN - 1);
  const int t = idx >> 11;          // token index (DIN = 2048)
  const int l = t & (LSEQ - 1);
  const int b = t >> 11;            // LSEQ = 2048
  float acc = conv_b[d];
#pragma unroll
  for (int j = 0; j < 4; ++j) {
    const int ls = l - 3 + j;
    if (ls >= 0)
      acc += x_raw[((size_t)(b * LSEQ + ls)) * DIN + d] * conv_w[d * 4 + j];
  }
  x_act[idx] = (__bf16)siluf(acc);
}

// ---------------- extract dt_low (cols 0..63 of x_dbl) as bf16 ----------------
__global__ void k_dtlow(const float* __restrict__ x_dbl, __bf16* __restrict__ dtlow)
{
  int i = blockIdx.x * 256 + threadIdx.x;
  if (i >= TTOK * RNK) return;
  const int t = i >> 6;
  const int r = i & 63;
  dtlow[i] = (__bf16)x_dbl[(size_t)t * XDBL + r];
}

// ---------------- fused selective scan + D-skip + SiLU(z) gate -> bf16 ----------------
__global__ __launch_bounds__(256)
void k_scan(const float* __restrict__ delta, const __bf16* __restrict__ x_act,
            const float* __restrict__ z_raw, const float* __restrict__ x_dbl,
            const float* __restrict__ A_log, const float* __restrict__ Dvec,
            __bf16* __restrict__ y_out)
{
  const int g = blockIdx.x * 256 + threadIdx.x;   // 0..4095
  const int d = g & (DIN - 1);
  const int b = g >> 11;

  float a[DSTATE], h[DSTATE];
#pragma unroll
  for (int n = 0; n < DSTATE; ++n) {
    a[n] = -__expf(A_log[d * DSTATE + n]);
    h[n] = 0.0f;
  }
  const float Dd = Dvec[d];

  for (int l = 0; l < LSEQ; ++l) {
    const size_t t  = (size_t)(b * LSEQ + l);
    const float dt  = delta[t * DIN + d];
    const float xv  = (float)x_act[t * DIN + d];
    const float zv  = z_raw[t * DIN + d];
    const float4* Bp = (const float4*)(x_dbl + t * XDBL + RNK);          // cols 64..79
    const float4* Cp = (const float4*)(x_dbl + t * XDBL + RNK + DSTATE); // cols 80..95
    const float dtx = dt * xv;
    float y = 0.0f;
#pragma unroll
    for (int q = 0; q < 4; ++q) {
      const float4 Bq = Bp[q];
      const float4 Cq = Cp[q];
      const float Bv[4] = {Bq.x, Bq.y, Bq.z, Bq.w};
      const float Cv[4] = {Cq.x, Cq.y, Cq.z, Cq.w};
#pragma unroll
      for (int i = 0; i < 4; ++i) {
        const int n = q * 4 + i;
        h[n] = __expf(dt * a[n]) * h[n] + dtx * Bv[i];
        y   += h[n] * Cv[i];
      }
    }
    y += Dd * xv;
    y *= siluf(zv);
    y_out[t * DIN + d] = (__bf16)y;
  }
}

extern "C" void kernel_launch(void* const* d_in, const int* in_sizes, int n_in,
                              void* d_out, int out_size, void* d_ws, size_t ws_size,
                              hipStream_t stream) {
  const float* x_norm     = (const float*)d_in[0];
  const float* in_proj_w  = (const float*)d_in[1];
  const float* conv_w     = (const float*)d_in[2];
  const float* conv_b     = (const float*)d_in[3];
  const float* x_proj_w   = (const float*)d_in[4];
  const float* dt_proj_w  = (const float*)d_in[5];
  const float* dt_proj_b  = (const float*)d_in[6];
  const float* out_proj_w = (const float*)d_in[7];
  const float* A_log      = (const float*)d_in[8];
  const float* Dvec       = (const float*)d_in[9];
  float* out = (float*)d_out;

  char* ws = (char*)d_ws;
  size_t off = 0;
  auto take = [&](size_t bytes) -> char* {
    char* p = ws + off;
    off = (off + bytes + 255) & ~(size_t)255;
    return p;
  };

  __bf16* xnorm_bf = (__bf16*)take((size_t)TTOK * DM * 2);
  __bf16* win_bf   = (__bf16*)take((size_t)2 * DIN * DM * 2);
  __bf16* wxp_bf   = (__bf16*)take((size_t)XDBL * DIN * 2);
  __bf16* wdt_bf   = (__bf16*)take((size_t)DIN * RNK * 2);
  __bf16* wout_bf  = (__bf16*)take((size_t)DM * DIN * 2);
  float*  x_raw    = (float*) take((size_t)TTOK * DIN * 4);
  float*  z_raw    = (float*) take((size_t)TTOK * DIN * 4);
  __bf16* x_act    = (__bf16*)take((size_t)TTOK * DIN * 2);
  float*  x_dbl    = (float*) take((size_t)TTOK * XDBL * 4);
  __bf16* dtlow    = (__bf16*)take((size_t)TTOK * RNK * 2);
  float*  delta    = (float*) take((size_t)TTOK * DIN * 4);
  __bf16* y_bf     = (__bf16*)take((size_t)TTOK * DIN * 2);

  auto cvt = [&](const float* src, __bf16* dst, int n) {
    k_cvt_bf16<<<(n + 255) / 256, 256, 0, stream>>>(src, dst, n);
  };
  cvt(x_norm,     xnorm_bf, TTOK * DM);
  cvt(in_proj_w,  win_bf,   2 * DIN * DM);
  cvt(x_proj_w,   wxp_bf,   XDBL * DIN);
  cvt(dt_proj_w,  wdt_bf,   DIN * RNK);
  cvt(out_proj_w, wout_bf,  DM * DIN);

  // 1) in_proj -> split x/z
  k_wmma_gemm<<<dim3((2 * DIN) / 64, TTOK / 64), 256, 0, stream>>>(
      xnorm_bf, win_bf, nullptr, x_raw, z_raw, 2 * DIN, DM, 1);

  // 2) causal depthwise conv + SiLU
  k_conv_silu<<<(TTOK * DIN) / 256, 256, 0, stream>>>(x_raw, conv_w, conv_b, x_act);

  // 3) x_proj -> x_dbl
  k_wmma_gemm<<<dim3((XDBL + 63) / 64, TTOK / 64), 256, 0, stream>>>(
      x_act, wxp_bf, nullptr, x_dbl, nullptr, XDBL, DIN, 0);

  // 4) dt_low -> bf16
  k_dtlow<<<(TTOK * RNK) / 256, 256, 0, stream>>>(x_dbl, dtlow);

  // 5) dt_proj + bias + softplus -> delta
  k_wmma_gemm<<<dim3(DIN / 64, TTOK / 64), 256, 0, stream>>>(
      dtlow, wdt_bf, dt_proj_b, delta, nullptr, DIN, RNK, 2);

  // 6) fused selective scan + skip + gate -> y
  k_scan<<<(2 * DIN) / 256, 256, 0, stream>>>(
      delta, x_act, z_raw, x_dbl, A_log, Dvec, y_bf);

  // 7) out_proj -> d_out
  k_wmma_gemm<<<dim3(DM / 64, TTOK / 64), 256, 0, stream>>>(
      y_bf, wout_bf, nullptr, out, nullptr, DM, DIN, 0);
}
